// AttentionBlock_16673063043126
// MI455X (gfx1250) — compile-verified, matrix operand order checked by reading, not run on hardware
//
#include <hip/hip_runtime.h>
#include <hip/hip_bf16.h>

#define BATCH 16
#define CH 512
#define NTOK 1024           // 32*32 spatial
#define HEADS 4
#define HD 128
#define NGROUPS 32

typedef __attribute__((ext_vector_type(16))) _Float16 v16h;
typedef __attribute__((ext_vector_type(8)))  float    v8f;
typedef unsigned int u32x4 __attribute__((ext_vector_type(4)));
typedef int          i32x4 __attribute__((ext_vector_type(4)));
typedef int          i32x8 __attribute__((ext_vector_type(8)));

union Frag16 { v16h v; int4 q[2]; };

// Load a 16x32 f16 WMMA operand fragment from a row-major tile (leading dim ld,
// in elements). For the A operand "row" = M index; for the B operand "row" = N
// (column of B), since B is fed per-column. Lane l<16: row l, K {ko..ko+7,
// ko+16..ko+23}; lanes 16-31: ko = 8.
__device__ inline v16h load_frag(const _Float16* __restrict__ p, int ld) {
    const int lane = threadIdx.x & 31;
    const int row  = lane & 15;
    const int ko   = (lane >> 4) << 3;
    const _Float16* b = p + (size_t)row * ld + ko;
    Frag16 f;
    f.q[0] = *reinterpret_cast<const int4*>(b);        // K ko..ko+7
    f.q[1] = *reinterpret_cast<const int4*>(b + 16);   // K ko+16..ko+23
    return f.v;
}

__device__ inline v8f wmma_f16(v16h a, v16h b, v8f c) {
    return __builtin_amdgcn_wmma_f32_16x16x32_f16(false, a, false, b,
                                                  (short)0, c, false, false);
}

#if __has_builtin(__builtin_amdgcn_tensor_load_to_lds)
#define HAVE_TDM 1
// Issue a TDM 2D tile load (f16 elements) global -> LDS.  D# layout per CDNA5
// ISA ch.8: group0 = {flags, lds_addr, global_addr, type=2}; group1 packs
// data_size / tensor dims / tile dims / dim0-stride.  All args are
// wave-uniform here (derived from blockIdx + loop counter).
__device__ inline void tdm_load_2d(unsigned lds_addr, const void* gaddr,
                                   unsigned td0, unsigned td1,
                                   unsigned tile0, unsigned tile1,
                                   unsigned long long stride0) {
    const unsigned long long ga = (unsigned long long)(uintptr_t)gaddr;
    u32x4 g0;
    g0[0] = 1u;                                        // count=1 (valid user D#)
    g0[1] = lds_addr;                                  // LDS byte address
    g0[2] = (unsigned)(ga & 0xFFFFFFFFu);              // global_addr[31:0]
    g0[3] = (unsigned)((ga >> 32) & 0x01FFFFFFu) | (2u << 30);  // [120:96] | type=2
    i32x8 g1;
    g1[0] = (int)(1u << 16);                           // data_size=1 (2B), wg_mask=0
    g1[1] = (int)((td0 & 0xFFFFu) << 16);              // tensor_dim0[15:0]
    g1[2] = (int)((td0 >> 16) | ((td1 & 0xFFFFu) << 16));
    g1[3] = (int)((td1 >> 16) | (tile0 << 16));        // tile_dim0
    g1[4] = (int)(tile1);                              // tile_dim1 (tile_dim2=0)
    g1[5] = (int)(unsigned)(stride0 & 0xFFFFFFFFull);  // tensor_dim0_stride lo
    g1[6] = (int)(unsigned)((stride0 >> 32) & 0xFFFFull);
    g1[7] = 0;
    const i32x4 gz = {0, 0, 0, 0};                     // 2D tile: groups 2/3 unused
#if defined(__clang_major__) && __clang_major__ >= 23
    const i32x8 gz8 = {0, 0, 0, 0, 0, 0, 0, 0};
    __builtin_amdgcn_tensor_load_to_lds(g0, g1, gz, gz, gz8, 0);
#else
    __builtin_amdgcn_tensor_load_to_lds(g0, g1, gz, gz, 0);
#endif
}
#else
#define HAVE_TDM 0
#endif

// ---------------------------------------------------------------------------
// Kernel 1: GroupNorm -> f16 activations in [b][n][c] layout
// ---------------------------------------------------------------------------
__global__ __launch_bounds__(256) void groupnorm_kernel(
    const float* __restrict__ x, const float* __restrict__ nw,
    const float* __restrict__ nb, _Float16* __restrict__ hbuf) {
    const int bg = blockIdx.x;            // 16*32 = 512 groups
    const int b = bg >> 5, g = bg & 31;
    const float* xp = x + (size_t)(b * CH + g * 16) * NTOK;  // 16384 contiguous floats

    float s = 0.f, ss = 0.f;
    for (int i = threadIdx.x; i < 4096; i += 256) {
        float4 v = reinterpret_cast<const float4*>(xp)[i];
        s  += v.x + v.y + v.z + v.w;
        ss += v.x * v.x + v.y * v.y + v.z * v.z + v.w * v.w;
    }
    __shared__ float red0[256], red1[256];
    red0[threadIdx.x] = s; red1[threadIdx.x] = ss;
    __syncthreads();
    for (int off = 128; off > 0; off >>= 1) {
        if (threadIdx.x < off) {
            red0[threadIdx.x] += red0[threadIdx.x + off];
            red1[threadIdx.x] += red1[threadIdx.x + off];
        }
        __syncthreads();
    }
    const float mean = red0[0] * (1.f / 16384.f);
    const float var  = red1[0] * (1.f / 16384.f) - mean * mean;
    const float rstd = rsqrtf(var + 1e-5f);

    for (int i = threadIdx.x; i < 16384; i += 256) {
        const int cc = i >> 10, n = i & 1023;
        const int cg = g * 16 + cc;
        float v = (xp[i] - mean) * rstd * nw[cg] + nb[cg];
        hbuf[((size_t)b * NTOK + n) * CH + cg] = (_Float16)v;
    }
}

// ---------------------------------------------------------------------------
// Kernel 2: weights f32 -> f16
// ---------------------------------------------------------------------------
__global__ __launch_bounds__(256) void wconvert_kernel(
    const float* __restrict__ qkv_w, const float* __restrict__ proj_w,
    _Float16* __restrict__ wq, _Float16* __restrict__ wp) {
    const int i = blockIdx.x * 256 + threadIdx.x;
    const int n1 = 3 * CH * CH;
    if (i < n1) wq[i] = (_Float16)qkv_w[i];
    const int j = i - n1;
    if (j >= 0 && j < CH * CH) wp[j] = (_Float16)proj_w[j];
}

// ---------------------------------------------------------------------------
// Kernel 3: QKV GEMM (per batch: [1536x512] x [512x1024]).
// Block = 8 waves, wave tile 32(M) x 64(N), block tile 128x128.
// Writes q (pre-scaled) & k as [b][h][n][hd], v as [b][h][hd][n].
// ---------------------------------------------------------------------------
__global__ __launch_bounds__(256) void qkv_gemm_kernel(
    const _Float16* __restrict__ w16, const float* __restrict__ bias,
    const _Float16* __restrict__ hbuf,
    _Float16* __restrict__ qb, _Float16* __restrict__ kb,
    _Float16* __restrict__ vb) {
    const int b    = blockIdx.z;
    const int wave = threadIdx.x >> 5;
    const int lane = threadIdx.x & 31;
    const int m0   = blockIdx.y * 128 + (wave >> 1) * 32;   // output row (o)
    const int n0   = blockIdx.x * 128 + (wave & 1) * 64;    // token (n)
    const _Float16* hb = hbuf + (size_t)b * NTOK * CH;

    v8f acc[2][4] = {};
    for (int kc = 0; kc < CH; kc += 32) {
        v16h a0 = load_frag(w16 + (size_t)m0 * CH + kc, CH);
        v16h a1 = load_frag(w16 + (size_t)(m0 + 16) * CH + kc, CH);
        #pragma unroll
        for (int j = 0; j < 4; ++j) {
            v16h bf = load_frag(hb + (size_t)(n0 + j * 16) * CH + kc, CH);
            acc[0][j] = wmma_f16(a0, bf, acc[0][j]);
            acc[1][j] = wmma_f16(a1, bf, acc[1][j]);
        }
    }

    const int colhi = lane >> 4, col = lane & 15;
    const float scale = 0.08838834764831845f;   // hd^-0.5
    #pragma unroll
    for (int mi = 0; mi < 2; ++mi)
        #pragma unroll
        for (int j = 0; j < 4; ++j)
            #pragma unroll
            for (int r = 0; r < 8; ++r) {
                const int o = m0 + mi * 16 + r + 8 * colhi;
                const int n = n0 + j * 16 + col;
                float v = acc[mi][j][r] + bias[o];
                const int sec = o >> 9;            // 0=q 1=k 2=v
                const int oo  = o & 511;
                const int h   = oo >> 7, c = oo & 127;
                const size_t bh = (size_t)(b * HEADS + h);
                if (sec == 0)      qb[(bh * NTOK + n) * HD + c] = (_Float16)(v * scale);
                else if (sec == 1) kb[(bh * NTOK + n) * HD + c] = (_Float16)v;
                else               vb[(bh * HD + c) * NTOK + n] = (_Float16)v;
            }
}

// ---------------------------------------------------------------------------
// Kernel 4: flash attention. One wave per 16 query rows; block = 4 waves.
// K/V tiles are DMA'd into LDS once per block by the Tensor Data Mover and
// shared by all 4 waves (4x traffic reduction vs per-wave global streams).
// Per 32-key step: 8 WMMA (S) + online softmax + LDS relayout + 8 WMMA (P*V).
// ---------------------------------------------------------------------------
__global__ __launch_bounds__(128) void attn_kernel(
    const _Float16* __restrict__ qb, const _Float16* __restrict__ kb,
    const _Float16* __restrict__ vb, _Float16* __restrict__ ob) {
    const int b = blockIdx.z, h = blockIdx.y;
    const int wave = threadIdx.x >> 5, lane = threadIdx.x & 31;
    const int n0 = (blockIdx.x * 4 + wave) * 16;
    const size_t bh = (size_t)(b * HEADS + h);
    const _Float16* qp = qb + (bh * NTOK + n0) * HD;
    const _Float16* kp = kb + bh * NTOK * HD;
    const _Float16* vp = vb + bh * HD * NTOK;

    __shared__ _Float16 klds[32 * HD];     // K tile: [m=32][c=128]
    __shared__ _Float16 vlds[HD * 32];     // V tile: [c=128][m=32]
    __shared__ _Float16 plds[4][16 * 32];  // per-wave P relayout slab
    _Float16* pl = &plds[wave][0];

    v16h aq[4];
    #pragma unroll
    for (int j = 0; j < 4; ++j) aq[j] = load_frag(qp + j * 32, HD);

    v8f o[8] = {};
    float rmax[8], rsum[8];
    #pragma unroll
    for (int r = 0; r < 8; ++r) { rmax[r] = -1e30f; rsum[r] = 0.f; }

    const int colhi = lane >> 4, col = lane & 15;

    for (int m0 = 0; m0 < NTOK; m0 += 32) {
        __syncthreads();   // previous tile fully consumed by all waves
#if HAVE_TDM
        if (wave == 0) {
            // K tile: 32 contiguous rows of 128 f16 (stride HD)
            tdm_load_2d((unsigned)(uintptr_t)(void*)klds, kp + (size_t)m0 * HD,
                        HD, 32, HD, 32, (unsigned long long)HD);
            // V tile: 128 rows of 32 f16, row stride NTOK
            tdm_load_2d((unsigned)(uintptr_t)(void*)vlds, vp + m0,
                        32, HD, 32, HD, (unsigned long long)NTOK);
            __builtin_amdgcn_s_wait_tensorcnt(0);
        }
#else
        // cooperative staged copy fallback
        for (int i = threadIdx.x; i < (32 * HD) / 8; i += 128)
            reinterpret_cast<int4*>(klds)[i] =
                reinterpret_cast<const int4*>(kp + (size_t)m0 * HD)[i];
        for (int i = threadIdx.x; i < (HD * 32) / 8; i += 128) {
            const int c = i >> 2, mm = (i & 3) * 8;
            reinterpret_cast<int4*>(vlds)[i] =
                *reinterpret_cast<const int4*>(vp + (size_t)c * NTOK + m0 + mm);
        }
#endif
        __syncthreads();   // tiles visible to all waves

        v8f s0 = {}, s1 = {};
        #pragma unroll
        for (int j = 0; j < 4; ++j) {
            v16h bk0 = load_frag(klds + j * 32, HD);
            v16h bk1 = load_frag(klds + 16 * HD + j * 32, HD);
            s0 = wmma_f16(aq[j], bk0, s0);
            s1 = wmma_f16(aq[j], bk1, s1);
        }
        // online softmax: rows of the D tile live across 16-lane groups
        float alpha[8];
        #pragma unroll
        for (int r = 0; r < 8; ++r) {
            float mx = fmaxf(s0[r], s1[r]);
            #pragma unroll
            for (int d = 1; d < 16; d <<= 1) mx = fmaxf(mx, __shfl_xor(mx, d, 32));
            const float nm = fmaxf(rmax[r], mx);
            alpha[r] = __expf(rmax[r] - nm);
            rmax[r] = nm;
            const float p0 = __expf(s0[r] - nm);
            const float p1 = __expf(s1[r] - nm);
            s0[r] = p0; s1[r] = p1;
            float ps = p0 + p1;
            #pragma unroll
            for (int d = 1; d < 16; d <<= 1) ps += __shfl_xor(ps, d, 32);
            rsum[r] = rsum[r] * alpha[r] + ps;
        }
        #pragma unroll
        for (int t = 0; t < 8; ++t)
            #pragma unroll
            for (int r = 0; r < 8; ++r) o[t][r] *= alpha[r];

        // P: D-layout -> row-major [16][32] in LDS -> A fragment
        #pragma unroll
        for (int r = 0; r < 8; ++r) {
            const int row = r + 8 * colhi;
            pl[row * 32 + col]      = (_Float16)s0[r];
            pl[row * 32 + 16 + col] = (_Float16)s1[r];
        }
        asm volatile("s_wait_dscnt 0" ::: "memory");
        v16h pa = load_frag(pl, 32);

        #pragma unroll
        for (int t = 0; t < 8; ++t) {
            v16h bv = load_frag(vlds + (size_t)(t * 16) * 32, 32);
            o[t] = wmma_f16(pa, bv, o[t]);
        }
        asm volatile("s_wait_dscnt 0" ::: "memory");  // pa consumed before next store
    }

    float inv[8];
    #pragma unroll
    for (int r = 0; r < 8; ++r) inv[r] = 1.f / rsum[r];
    #pragma unroll
    for (int t = 0; t < 8; ++t)
        #pragma unroll
        for (int r = 0; r < 8; ++r) {
            const int row = n0 + r + 8 * colhi;
            const int cg  = h * HD + t * 16 + col;
            ob[((size_t)b * NTOK + row) * CH + cg] = (_Float16)(o[t][r] * inv[r]);
        }
}

// ---------------------------------------------------------------------------
// Kernel 5: proj GEMM + bias + residual -> f32 output
// ---------------------------------------------------------------------------
__global__ __launch_bounds__(256) void proj_gemm_kernel(
    const _Float16* __restrict__ w16, const float* __restrict__ bias,
    const _Float16* __restrict__ ob, const float* __restrict__ x,
    float* __restrict__ out) {
    const int b    = blockIdx.z;
    const int wave = threadIdx.x >> 5;
    const int lane = threadIdx.x & 31;
    const int m0   = blockIdx.y * 128 + (wave >> 1) * 32;
    const int n0   = blockIdx.x * 128 + (wave & 1) * 64;
    const _Float16* obb = ob + (size_t)b * NTOK * CH;

    v8f acc[2][4] = {};
    for (int kc = 0; kc < CH; kc += 32) {
        v16h a0 = load_frag(w16 + (size_t)m0 * CH + kc, CH);
        v16h a1 = load_frag(w16 + (size_t)(m0 + 16) * CH + kc, CH);
        #pragma unroll
        for (int j = 0; j < 4; ++j) {
            v16h bf = load_frag(obb + (size_t)(n0 + j * 16) * CH + kc, CH);
            acc[0][j] = wmma_f16(a0, bf, acc[0][j]);
            acc[1][j] = wmma_f16(a1, bf, acc[1][j]);
        }
    }
    const int colhi = lane >> 4, col = lane & 15;
    #pragma unroll
    for (int mi = 0; mi < 2; ++mi)
        #pragma unroll
        for (int j = 0; j < 4; ++j)
            #pragma unroll
            for (int r = 0; r < 8; ++r) {
                const int o = m0 + mi * 16 + r + 8 * colhi;
                const int n = n0 + j * 16 + col;
                const size_t idx = ((size_t)b * CH + o) * NTOK + n;
                out[idx] = x[idx] + acc[mi][j][r] + bias[o];
            }
}

// ---------------------------------------------------------------------------
extern "C" void kernel_launch(void* const* d_in, const int* in_sizes, int n_in,
                              void* d_out, int out_size, void* d_ws, size_t ws_size,
                              hipStream_t stream) {
    const float* x      = (const float*)d_in[0];
    const float* norm_w = (const float*)d_in[1];
    const float* norm_b = (const float*)d_in[2];
    const float* qkv_w  = (const float*)d_in[3];
    const float* qkv_b  = (const float*)d_in[4];
    const float* proj_w = (const float*)d_in[5];
    const float* proj_b = (const float*)d_in[6];
    float* out = (float*)d_out;

    char* ws = (char*)d_ws;
    const size_t ACT = (size_t)BATCH * NTOK * CH * sizeof(_Float16);   // 16 MB
    _Float16* hbuf = (_Float16*)(ws + 0 * ACT);
    _Float16* qb   = (_Float16*)(ws + 1 * ACT);
    _Float16* kb   = (_Float16*)(ws + 2 * ACT);
    _Float16* vb   = (_Float16*)(ws + 3 * ACT);
    _Float16* obuf = (_Float16*)(ws + 4 * ACT);
    _Float16* wq16 = (_Float16*)(ws + 5 * ACT);
    _Float16* wp16 = (_Float16*)(ws + 5 * ACT + (size_t)3 * CH * CH * sizeof(_Float16));

    wconvert_kernel<<<(3 * CH * CH + CH * CH + 255) / 256, 256, 0, stream>>>(
        qkv_w, proj_w, wq16, wp16);
    groupnorm_kernel<<<BATCH * NGROUPS, 256, 0, stream>>>(x, norm_w, norm_b, hbuf);
    qkv_gemm_kernel<<<dim3(NTOK / 128, 3 * CH / 128, BATCH), 256, 0, stream>>>(
        wq16, qkv_b, hbuf, qb, kb, vb);
    attn_kernel<<<dim3(NTOK / 64, HEADS, BATCH), 128, 0, stream>>>(qb, kb, vb, obuf);
    proj_gemm_kernel<<<dim3(NTOK / 128, CH / 128, BATCH), 256, 0, stream>>>(
        wp16, proj_b, obuf, x, out);
}
